// TopKSAE_10849087389716
// MI455X (gfx1250) — compile-verified, matrix operand order checked by reading, not run on hardware
//
#include <hip/hip_runtime.h>
#include <hip/hip_bf16.h>
#include <stdint.h>

typedef __bf16 bf16_t;
typedef __attribute__((ext_vector_type(16))) __bf16       v16bf;
typedef __attribute__((ext_vector_type(8)))  float        v8f;
typedef __attribute__((ext_vector_type(4)))  float        v4f;
typedef __attribute__((ext_vector_type(4)))  unsigned int v4u;
typedef __attribute__((ext_vector_type(8)))  unsigned int v8u;

#define ROWS 12608            // 64*197
#define DVIT 768
#define DSAE 12288
#define TOPK 32
#define NCHUNKS 6
#define NCHUNK 2048           // DSAE / NCHUNKS
#define MROWS 32
#define WAVES1 16
#define BLK1 (WAVES1 * 32)
#define KSTEPS (DVIT / 32)    // 24
#define CT_PER_WAVE (NCHUNK / 16 / WAVES1) // 8
#define ACT_STRIDE 2049       // dwords (pad to break bank alignment)
#define A_STRIDE 776          // halves (768 + 8 pad)
#define CAND_PER_ROW (NCHUNKS * TOPK)      // 192

// ---------------- Kernel 0a: W_enc[768,12288] f32 -> W_enc^T[12288,768] bf16 ----
__global__ void __launch_bounds__(256)
wencT_bf16_kernel(const float* __restrict__ W_enc, bf16_t* __restrict__ out) {
    __shared__ float tile[32][33];
    const int nb = blockIdx.x * 32;
    const int kb = blockIdx.y * 32;
    const int tx = threadIdx.x, ty = threadIdx.y; // 32 x 8
#pragma unroll
    for (int i = 0; i < 4; ++i) {
        int k = ty + i * 8;
        tile[k][tx] = W_enc[(size_t)(kb + k) * DSAE + nb + tx];
    }
    __syncthreads();
#pragma unroll
    for (int i = 0; i < 4; ++i) {
        int n = ty + i * 8;
        out[(size_t)(nb + n) * DVIT + kb + tx] = (bf16_t)tile[tx][n];
    }
}

// ---------------- Kernel 0b: xc = bf16(x - b_dec) -------------------------------
__global__ void __launch_bounds__(256)
xcent_bf16_kernel(const float* __restrict__ x, const float* __restrict__ b_dec,
                  bf16_t* __restrict__ xc) {
    const int i = blockIdx.x * 256 + threadIdx.x;
    const int c = (blockIdx.x % 3) * 256 + threadIdx.x; // 768 = 3*256
    xc[i] = (bf16_t)(x[i] - b_dec[c]);
}

// ---------------- Kernel 1: fused encoder GEMM (bf16 WMMA) + per-chunk top-32 ---
__global__ void __launch_bounds__(BLK1)
encoder_topk_kernel(const bf16_t* __restrict__ xc, const bf16_t* __restrict__ wt,
                    float* __restrict__ candVal, int* __restrict__ candIdx) {
    extern __shared__ unsigned char smem[];
    float*        acts = (float*)smem;                                   // 32*2049 f32
    unsigned int* aT   = (unsigned int*)(smem + (size_t)MROWS * ACT_STRIDE * 4);

    const int tid  = threadIdx.x;
    const int lane = tid & 31;
    const int wv   = tid >> 5;
    const int rowBlk = blockIdx.x;
    const int chunk  = blockIdx.y;

    // Stage A: 32 rows x 768 halves of xc -> LDS via CDNA5 async copies
    // (global_load_async_to_lds_b128, tracked on ASYNCcnt; each thread moves 96B)
    {
        const unsigned char* src =
            (const unsigned char*)xc + (size_t)rowBlk * MROWS * DVIT * 2;
        const int r  = tid >> 4;           // 0..31 (row)
        const int c0 = (tid & 15) * 96;    // byte column within row
        const unsigned char* g = src + (size_t)r * (DVIT * 2) + c0;
        // generic->LDS: low 32 bits of the generic pointer are the LDS byte offset
        unsigned lds0 =
            (unsigned)(uintptr_t)((unsigned char*)aT + r * (A_STRIDE * 2) + c0);
#pragma unroll
        for (int j = 0; j < 6; ++j) {
            unsigned long long ga = (unsigned long long)(uintptr_t)(g + j * 16);
            unsigned la = lds0 + j * 16;
            asm volatile("global_load_async_to_lds_b128 %0, %1, off"
                         :: "v"(la), "v"(ga) : "memory");
        }
        asm volatile("s_wait_asynccnt 0x0" ::: "memory");
    }
    __syncthreads();

    const int ln = lane & 15;
    const int kg = (lane >> 4) * 8;  // per-lane K half-group offset (WMMA layout)

    // Phase 1: each wave computes 8 column tiles x 2 row tiles over K=768
    for (int ct = 0; ct < CT_PER_WAVE; ++ct) {
        const int ctCol = (wv * CT_PER_WAVE + ct) * 16;          // col within chunk
        const int nGlob = chunk * NCHUNK + ctCol + ln;           // feature index
        const bf16_t* bbase = wt + (size_t)nGlob * DVIT + kg;
        // prefetch next column tile's B rows (global_prefetch_b8) to keep the
        // L2 stream warm across the clause-pipeline restart at tile boundaries
        if (ct + 1 < CT_PER_WAVE) {
            __builtin_prefetch(bbase + (size_t)16 * DVIT, 0, 0);
        }
        v8f acc0 = {};
        v8f acc1 = {};
        for (int ks = 0; ks < KSTEPS; ++ks) {
            const int kb = ks * 32;
            // B operand: two b128 loads of K-contiguous bf16 from W_enc^T
            v4u b0 = *(const v4u*)(bbase + kb);
            v4u b1 = *(const v4u*)(bbase + kb + 16);
            v8u bw; bw.lo = b0; bw.hi = b1;
            v16bf vb = __builtin_bit_cast(v16bf, bw);
            // A operands from LDS (row tile 0 and +16)
            const unsigned int* a0p =
                aT + (size_t)ln * (A_STRIDE / 2) + (kb + kg) / 2;
            v4u a00 = *(const v4u*)a0p;
            v4u a01 = *(const v4u*)(a0p + 8);
            v8u aw0; aw0.lo = a00; aw0.hi = a01;
            v16bf va0 = __builtin_bit_cast(v16bf, aw0);
            const unsigned int* a1p = a0p + 16 * (A_STRIDE / 2);
            v4u a10 = *(const v4u*)a1p;
            v4u a11 = *(const v4u*)(a1p + 8);
            v8u aw1; aw1.lo = a10; aw1.hi = a11;
            v16bf va1 = __builtin_bit_cast(v16bf, aw1);

            acc0 = __builtin_amdgcn_wmma_f32_16x16x32_bf16(
                false, va0, false, vb, (short)0, acc0, false, false);
            acc1 = __builtin_amdgcn_wmma_f32_16x16x32_bf16(
                false, va1, false, vb, (short)0, acc1, false, false);
        }
        // ReLU + store C tiles (lane 0-15: rows 0-7; lane 16-31: rows 8-15)
        const int rb  = (lane >> 4) * 8;
        const int col = ctCol + ln;
#pragma unroll
        for (int v = 0; v < 8; ++v) {
            float f0 = acc0[v]; f0 = f0 > 0.f ? f0 : 0.f;
            float f1 = acc1[v]; f1 = f1 > 0.f ? f1 : 0.f;
            acts[(size_t)(rb + v) * ACT_STRIDE + col]      = f0;
            acts[(size_t)(16 + rb + v) * ACT_STRIDE + col] = f1;
        }
    }
    __syncthreads();

    // Phase 2: per-row top-32 within this 2048-col chunk (one wave per 2 rows)
    for (int rr = 0; rr < 2; ++rr) {
        const int row = wv * 2 + rr;
        float vals[64];
#pragma unroll
        for (int j = 0; j < 64; ++j)
            vals[j] = acts[(size_t)row * ACT_STRIDE + lane + 32 * j];
        unsigned long long used = 0ull;
        float selV = 0.f; int selI = 0;
        for (int t = 0; t < TOPK; ++t) {
            // per-lane masked argmax over 64 register-resident values (>= 0)
            float best = -1.f; int bj = 0;
#pragma unroll
            for (int j = 0; j < 64; ++j) {
                bool tk = (((used >> j) & 1ull) == 0ull) && (vals[j] > best);
                best = tk ? vals[j] : best;
                bj   = tk ? j : bj;
            }
            // wave argmax butterfly with a consistent (val, lane) comparator
            float bv = best; int bl = lane; int bjj = bj;
#pragma unroll
            for (int off = 16; off > 0; off >>= 1) {
                float ov = __shfl_xor(bv, off, 32);
                int   ol = __shfl_xor(bl, off, 32);
                int   oj = __shfl_xor(bjj, off, 32);
                bool  tk = (ov > bv) || (ov == bv && ol < bl);
                bv = tk ? ov : bv; bl = tk ? ol : bl; bjj = tk ? oj : bjj;
            }
            if (lane == bl) used |= (1ull << bjj);
            if (lane == t) { selV = bv; selI = chunk * NCHUNK + bl + 32 * bjj; }
        }
        const int gRow = rowBlk * MROWS + row;
        candVal[(size_t)gRow * CAND_PER_ROW + chunk * TOPK + lane] = selV;
        candIdx[(size_t)gRow * CAND_PER_ROW + chunk * TOPK + lane] = selI;
    }
}

// ---------------- Kernel 2: merge 192 candidates -> top-32, sparse decode -------
__global__ void __launch_bounds__(256)
merge_decode_kernel(const float* __restrict__ candVal, const int* __restrict__ candIdx,
                    const float* __restrict__ W_dec, const float* __restrict__ b_dec,
                    float* __restrict__ out) {
    const int lane = threadIdx.x & 31;
    const int wv   = threadIdx.x >> 5;
    const int row  = blockIdx.x * 8 + wv;   // 12608 = 1576 * 8

    float vals[6]; int idxs[6];
#pragma unroll
    for (int j = 0; j < 6; ++j) {
        vals[j] = candVal[(size_t)row * CAND_PER_ROW + lane + 32 * j];
        idxs[j] = candIdx[(size_t)row * CAND_PER_ROW + lane + 32 * j];
    }
    unsigned int used = 0u;
    float selV = 0.f; int selI = 0;
    for (int t = 0; t < TOPK; ++t) {
        float best = -1.f; int bj = 0;
#pragma unroll
        for (int j = 0; j < 6; ++j) {
            bool tk = (((used >> j) & 1u) == 0u) && (vals[j] > best);
            best = tk ? vals[j] : best;
            bj   = tk ? j : bj;
        }
        float bv = best; int bl = lane; int bjj = bj;
#pragma unroll
        for (int off = 16; off > 0; off >>= 1) {
            float ov = __shfl_xor(bv, off, 32);
            int   ol = __shfl_xor(bl, off, 32);
            int   oj = __shfl_xor(bjj, off, 32);
            bool  tk = (ov > bv) || (ov == bv && ol < bl);
            bv = tk ? ov : bv; bl = tk ? ol : bl; bjj = tk ? oj : bjj;
        }
        // fetch winning candidate's feature index from lane bl, slot bjj
        int myIdx = idxs[0];
#pragma unroll
        for (int j = 1; j < 6; ++j) myIdx = (bjj == j) ? idxs[j] : myIdx;
        int wIdx = __shfl(myIdx, bl, 32);
        if (lane == bl) used |= (1u << bjj);
        if (lane == t) { selV = bv; selI = wIdx; }
    }
    // decode: x_hat[row,:] = b_dec + sum_i selV_i * W_dec[selI_i, :]
    v4f acc[6] = {};
    for (int t = 0; t < TOPK; ++t) {
        float v  = __shfl(selV, t, 32);
        int   ix = __shfl(selI, t, 32);
        const v4f* wrow = (const v4f*)(W_dec + (size_t)ix * DVIT);
#pragma unroll
        for (int j = 0; j < 6; ++j)
            acc[j] += v * wrow[lane + 32 * j];
    }
    const v4f* bd = (const v4f*)b_dec;
    v4f* o = (v4f*)(out + (size_t)row * DVIT);
#pragma unroll
    for (int j = 0; j < 6; ++j)
        o[lane + 32 * j] = acc[j] + bd[lane + 32 * j];
}

// ---------------- launch --------------------------------------------------------
extern "C" void kernel_launch(void* const* d_in, const int* in_sizes, int n_in,
                              void* d_out, int out_size, void* d_ws, size_t ws_size,
                              hipStream_t stream) {
    (void)in_sizes; (void)n_in; (void)out_size; (void)ws_size;
    const float* x     = (const float*)d_in[0];
    const float* W_enc = (const float*)d_in[1];
    const float* W_dec = (const float*)d_in[2];
    const float* b_dec = (const float*)d_in[3];
    float* out = (float*)d_out;

    unsigned char* ws = (unsigned char*)d_ws;
    const size_t OFF_WT   = 0;                                   // 12288*768*2
    const size_t OFF_XC   = OFF_WT + (size_t)DSAE * DVIT * 2;    // 12608*768*2
    const size_t OFF_CV   = OFF_XC + (size_t)ROWS * DVIT * 2;    // 12608*192*4
    const size_t OFF_CI   = OFF_CV + (size_t)ROWS * CAND_PER_ROW * 4;
    bf16_t* wt      = (bf16_t*)(ws + OFF_WT);
    bf16_t* xc      = (bf16_t*)(ws + OFF_XC);
    float*  candVal = (float*)(ws + OFF_CV);
    int*    candIdx = (int*)(ws + OFF_CI);

    wencT_bf16_kernel<<<dim3(DSAE / 32, DVIT / 32), dim3(32, 8), 0, stream>>>(W_enc, wt);
    xcent_bf16_kernel<<<dim3((ROWS * DVIT) / 256), dim3(256), 0, stream>>>(x, b_dec, xc);

    const size_t smem = (size_t)MROWS * ACT_STRIDE * 4 + (size_t)MROWS * A_STRIDE * 2;
    hipFuncSetAttribute((const void*)encoder_topk_kernel,
                        hipFuncAttributeMaxDynamicSharedMemorySize, (int)smem);
    encoder_topk_kernel<<<dim3(ROWS / MROWS, NCHUNKS), dim3(BLK1), smem, stream>>>(
        xc, wt, candVal, candIdx);

    merge_decode_kernel<<<dim3(ROWS / 8), dim3(256), 0, stream>>>(
        candVal, candIdx, W_dec, b_dec, out);
}
